// CompositeFullyConnected_57320633532772
// MI455X (gfx1250) — matrix operand order repaired and checked by reading, not compiled, not run on hardware
//
#include <hip/hip_runtime.h>
#include <hip/hip_bf16.h>

// ---------------- problem constants (from reference) ----------------
#define B_SAMPLES 4096
#define F_IN      512
#define H1_DIM    1024
#define H2_DIM    1024
#define U1_DIM    512
#define U2_DIM    256
#define K_BLOCKS  16

// ---------------- types ----------------
typedef __attribute__((ext_vector_type(16))) __bf16          v16bf;
typedef __attribute__((ext_vector_type(8)))  unsigned int    v8u;
typedef __attribute__((ext_vector_type(8)))  float           v8f;

__device__ __forceinline__ unsigned short f32_to_bf16_rne(float f) {
  unsigned int u = __float_as_uint(f);
  u += 0x7fffu + ((u >> 16) & 1u);          // round-to-nearest-even
  return (unsigned short)(u >> 16);
}

// ---------------- elementwise f32 -> bf16 (row-major kept: activations) ----
__global__ void cvt_f32_bf16(const float* __restrict__ in,
                             unsigned short* __restrict__ out, long n) {
  long i = (long)blockIdx.x * blockDim.x + threadIdx.x;
  if (i < n) out[i] = f32_to_bf16_rne(in[i]);
}

// dense weight [F][N] f32 -> transposed [N][F] bf16 (u-major for WMMA B side)
__global__ void pack_dense_T(const float* __restrict__ in,
                             unsigned short* __restrict__ out, int F, int N) {
  long total = (long)F * N;
  long o = (long)blockIdx.x * blockDim.x + threadIdx.x;
  if (o >= total) return;
  int f = (int)(o % F);
  int n = (int)(o / F);
  out[o] = f32_to_bf16_rne(in[(long)f * N + n]);
}

// blocked weight (f,u,k) f32 -> [k][u][f] bf16 (per-block, u-major)
__global__ void pack_blocked_T(const float* __restrict__ in,
                               unsigned short* __restrict__ out,
                               int F, int U, int K) {
  long total = (long)F * U * K;
  long o = (long)blockIdx.x * blockDim.x + threadIdx.x;
  if (o >= total) return;
  int f = (int)(o % F);
  long t = o / F;
  int u = (int)(t % U);
  int k = (int)(t / U);
  out[o] = f32_to_bf16_rne(in[((long)f * U + u) * K + k]);
}

// ---------------- bucket samples by automaton state ----------------
__global__ void bucket_by_state(const int* __restrict__ states, int nB,
                                int* __restrict__ perm,
                                int* __restrict__ offsets) {
  __shared__ int cnt[K_BLOCKS];
  __shared__ int cur[K_BLOCKS];
  const int tid = threadIdx.x;
  if (tid < K_BLOCKS) cnt[tid] = 0;
  __syncthreads();
  for (int i = tid; i < nB; i += blockDim.x)
    atomicAdd(&cnt[states[i] & (K_BLOCKS - 1)], 1);
  __syncthreads();
  if (tid == 0) {
    int run = 0;
    for (int k = 0; k < K_BLOCKS; ++k) {
      offsets[k] = run; cur[k] = run; run += cnt[k];
    }
    offsets[K_BLOCKS] = run;
  }
  __syncthreads();
  for (int i = tid; i < nB; i += blockDim.x) {
    int s = states[i] & (K_BLOCKS - 1);
    int p = atomicAdd(&cur[s], 1);
    perm[p] = i;
  }
}

// ---------------- bf16 WMMA GEMM, 128x128x64 tiles, 8 waves ----------------
// C[m,n] = relu?( sum_k A[m,k]*Bt[n,k] + bias[n] )   (B pre-transposed)
template <bool RELU, bool OUT_F32, bool GATHER_A, bool SCATTER_OUT>
__global__ __launch_bounds__(256)
void gemm_bf16_wmma(const unsigned short* __restrict__ A, int lda,
                    const unsigned short* __restrict__ Bt,  // [N][Kdim] bf16
                    int N, int Kdim,
                    const float* __restrict__ bias, int bstride,
                    void* __restrict__ Cout, int ldc,
                    const int* __restrict__ offsets,
                    const int* __restrict__ rowmap,
                    int Mtotal) {
  constexpr int BK  = 64;  // staged K depth (2 WMMA K-steps)
  constexpr int LDT = 72;  // 64 + 8 pad; row pitch 144B (16B aligned)
  __shared__ __attribute__((aligned(16))) unsigned short sA[128 * LDT];
  __shared__ __attribute__((aligned(16))) unsigned short sB[128 * LDT];

  const int tid  = threadIdx.x;
  const int lane = tid & 31;
  const int wave = tid >> 5;
  const int wm   = (wave & 3) * 32;   // wave M offset inside 128
  const int wn   = (wave >> 2) * 64;  // wave N offset inside 128

  int row_base = 0, mcount = Mtotal;
  const unsigned short* Bp = Bt;
  const float* bp = bias;
  if (offsets) {
    const int k = blockIdx.z;
    row_base = offsets[k];
    mcount   = offsets[k + 1] - row_base;
    Bp       = Bt + (size_t)k * N * Kdim;
    bp       = bias + k;                 // blocked bias layout: bias[u*K + k]
  }
  const int m0 = blockIdx.y * 128;
  if (m0 >= mcount) return;
  const int n0 = blockIdx.x * 128;

  // staging: 256 threads x 32 bf16 = one 128x64 tile (each matrix)
  const int e   = tid * 32;
  const int srl = e >> 6;       // 0..127 (A row / B n-row)
  const int scl = e & 63;       // 0 or 32 (K offset)

  const int  aridx  = m0 + srl;
  const bool avalid = aridx < mcount;
  long arow = 0;
  if (avalid)
    arow = GATHER_A ? (long)rowmap[row_base + aridx] : (long)(row_base + aridx);
  const unsigned short* aptr = A  + arow * (long)lda + scl;
  const unsigned short* bptr = Bp + (size_t)(n0 + srl) * Kdim + scl;

  v8f acc[2][4] = {};

  for (int kt = 0; kt < Kdim; kt += BK) {
    // -------- global -> regs (b128), zero-fill rows past bucket end
    uint4 ar[4] = {make_uint4(0,0,0,0), make_uint4(0,0,0,0),
                   make_uint4(0,0,0,0), make_uint4(0,0,0,0)};
    if (avalid) {
      const uint4* p = (const uint4*)(aptr + kt);
#pragma unroll
      for (int i = 0; i < 4; ++i) ar[i] = p[i];
    }
    uint4 br[4];
    {
      const uint4* q = (const uint4*)(bptr + kt);
#pragma unroll
      for (int i = 0; i < 4; ++i) br[i] = q[i];
    }
    if (kt + BK < Kdim) {  // gfx1250 global_prefetch_b8 on next K tile
      if (avalid) __builtin_prefetch(aptr + kt + BK, 0, 0);
      __builtin_prefetch(bptr + kt + BK, 0, 0);
    }
    __syncthreads();
    // -------- regs -> LDS (both tiles row-major along K, pitch LDT)
#pragma unroll
    for (int i = 0; i < 4; ++i) {
      *(uint4*)&sA[srl * LDT + scl + 8 * i] = ar[i];
      *(uint4*)&sB[srl * LDT + scl + 8 * i] = br[i];
    }
    __syncthreads();

#pragma unroll
    for (int kk = 0; kk < BK; kk += 32) {
      // A fragment (ISA 7.12.2, 16-bit A 16x32): per lane two contiguous
      // 8-element runs: K = [kh*8, +8) and [16+kh*8, +8)  -> 2x ds_load_b128
      v16bf afrag[2];
#pragma unroll
      for (int mt = 0; mt < 2; ++mt) {
        const int r   = wm + mt * 16 + (lane & 15);
        const int kh8 = (lane >> 4) * 8;
        const uint4 lo = *(const uint4*)&sA[r * LDT + kk + kh8];
        const uint4 hi = *(const uint4*)&sA[r * LDT + kk + 16 + kh8];
        v8u au;
        au[0] = lo.x; au[1] = lo.y; au[2] = lo.z; au[3] = lo.w;
        au[4] = hi.x; au[5] = hi.y; au[6] = hi.z; au[7] = hi.w;
        afrag[mt] = __builtin_bit_cast(v16bf, au);
      }
      // B fragment (16-bit B 32x16): lane=col n; per lane two 8-element runs:
      // K = [kh*8, +8) and [16+kh*8, +8) (same order as A) -> 2x b128
      v16bf bfrag[4];
#pragma unroll
      for (int nt = 0; nt < 4; ++nt) {
        const int c    = wn + nt * 16 + (lane & 15);
        const int kb0  = (lane >> 4) * 8;
        const uint4 lo = *(const uint4*)&sB[c * LDT + kk + kb0];
        const uint4 hi = *(const uint4*)&sB[c * LDT + kk + 16 + kb0];
        v8u bu;
        bu[0] = lo.x; bu[1] = lo.y; bu[2] = lo.z; bu[3] = lo.w;
        bu[4] = hi.x; bu[5] = hi.y; bu[6] = hi.z; bu[7] = hi.w;
        bfrag[nt] = __builtin_bit_cast(v16bf, bu);
      }
      // 8 WMMAs per K-step, 16 per staged tile
#pragma unroll
      for (int mt = 0; mt < 2; ++mt)
#pragma unroll
        for (int nt = 0; nt < 4; ++nt)
          acc[mt][nt] = __builtin_amdgcn_wmma_f32_16x16x32_bf16(
              false, afrag[mt], false, bfrag[nt],
              (short)0, acc[mt][nt], false, false);
    }
  }

  // -------- epilogue: bias + relu, store (C/D: VGPR i = M row i / 8+i)
  const int l15 = lane & 15;
  const int hi8 = (lane >> 4) * 8;
#pragma unroll
  for (int nt = 0; nt < 4; ++nt) {
    const int n  = n0 + wn + nt * 16 + l15;
    const float bv = bp[n * bstride];
#pragma unroll
    for (int mt = 0; mt < 2; ++mt) {
      const int rl = wm + mt * 16 + hi8;
#pragma unroll
      for (int i = 0; i < 8; ++i) {
        const int ridx = m0 + rl + i;
        if (ridx < mcount) {
          float v = acc[mt][nt][i] + bv;
          if (RELU) v = fmaxf(v, 0.0f);
          const long orow = SCATTER_OUT ? (long)rowmap[row_base + ridx]
                                        : (long)(row_base + ridx);
          if (OUT_F32)
            ((float*)Cout)[orow * (long)ldc + n] = v;
          else
            ((unsigned short*)Cout)[orow * (long)ldc + n] = f32_to_bf16_rne(v);
        }
      }
    }
  }
}

// ---------------- host launcher ----------------
extern "C" void kernel_launch(void* const* d_in, const int* in_sizes, int n_in,
                              void* d_out, int out_size, void* d_ws, size_t ws_size,
                              hipStream_t stream) {
  (void)in_sizes; (void)n_in; (void)out_size; (void)ws_size;
  const float* x   = (const float*)d_in[0];
  const float* W1  = (const float*)d_in[1];
  const float* b1  = (const float*)d_in[2];
  const float* W2  = (const float*)d_in[3];
  const float* b2  = (const float*)d_in[4];
  const float* Kb1 = (const float*)d_in[5];
  const float* Bb1 = (const float*)d_in[6];
  const float* Kb2 = (const float*)d_in[7];
  const float* Bb2 = (const float*)d_in[8];
  const int*   st  = (const int*)d_in[9];
  float* out = (float*)d_out;

  // workspace carve-up (256B aligned)
  char* ws = (char*)d_ws;
  size_t off = 0;
  auto carve = [&](size_t bytes) -> char* {
    char* p = ws + off;
    off = (off + bytes + 255) & ~(size_t)255;
    return p;
  };
  unsigned short* xb   = (unsigned short*)carve((size_t)B_SAMPLES * F_IN * 2);
  unsigned short* W1t  = (unsigned short*)carve((size_t)F_IN * H1_DIM * 2);   // [H1][F]
  unsigned short* W2t  = (unsigned short*)carve((size_t)H1_DIM * H2_DIM * 2); // [H2][H1]
  unsigned short* Kb1t = (unsigned short*)carve((size_t)K_BLOCKS * H2_DIM * U1_DIM * 2); // [k][U1][H2]
  unsigned short* Kb2t = (unsigned short*)carve((size_t)K_BLOCKS * U1_DIM * U2_DIM * 2); // [k][U2][U1]
  unsigned short* h1   = (unsigned short*)carve((size_t)B_SAMPLES * H1_DIM * 2);
  unsigned short* h2   = (unsigned short*)carve((size_t)B_SAMPLES * H2_DIM * 2);
  unsigned short* t1   = (unsigned short*)carve((size_t)B_SAMPLES * U1_DIM * 2);
  int* perm    = (int*)carve((size_t)B_SAMPLES * 4);
  int* offsets = (int*)carve((K_BLOCKS + 1) * 4);

  const int T = 256;
  auto grids = [&](long n) { return (unsigned)((n + T - 1) / T); };

  // 1) precision conversion + transposed weight repack
  cvt_f32_bf16 <<<grids((long)B_SAMPLES * F_IN), T, 0, stream>>>(x, xb, (long)B_SAMPLES * F_IN);
  pack_dense_T <<<grids((long)F_IN * H1_DIM),    T, 0, stream>>>(W1, W1t, F_IN,   H1_DIM);
  pack_dense_T <<<grids((long)H1_DIM * H2_DIM),  T, 0, stream>>>(W2, W2t, H1_DIM, H2_DIM);
  pack_blocked_T<<<grids((long)H2_DIM * U1_DIM * K_BLOCKS), T, 0, stream>>>(Kb1, Kb1t, H2_DIM, U1_DIM, K_BLOCKS);
  pack_blocked_T<<<grids((long)U1_DIM * U2_DIM * K_BLOCKS), T, 0, stream>>>(Kb2, Kb2t, U1_DIM, U2_DIM, K_BLOCKS);

  // 2) bucket samples by automaton state (16x pruning of blocked layers)
  bucket_by_state<<<1, 256, 0, stream>>>(st, B_SAMPLES, perm, offsets);

  // 3) shared dense layers: h1 = relu(x@W1+b1), h2 = relu(h1@W2+b2)
  gemm_bf16_wmma<true, false, false, false>
      <<<dim3(H1_DIM / 128, B_SAMPLES / 128, 1), 256, 0, stream>>>(
          xb, F_IN, W1t, H1_DIM, F_IN, b1, 1, h1, H1_DIM, nullptr, nullptr, B_SAMPLES);
  gemm_bf16_wmma<true, false, false, false>
      <<<dim3(H2_DIM / 128, B_SAMPLES / 128, 1), 256, 0, stream>>>(
          h1, H1_DIM, W2t, H2_DIM, H1_DIM, b2, 1, h2, H2_DIM, nullptr, nullptr, B_SAMPLES);

  // 4) blocked layer 1: rows gathered via perm, output in permuted row space
  gemm_bf16_wmma<true, false, true, false>
      <<<dim3(U1_DIM / 128, B_SAMPLES / 128, K_BLOCKS), 256, 0, stream>>>(
          h2, H2_DIM, Kb1t, U1_DIM, H2_DIM, Bb1, K_BLOCKS, t1, U1_DIM, offsets, perm, B_SAMPLES);

  // 5) blocked output layer: contiguous permuted rows in, scatter f32 out
  gemm_bf16_wmma<false, true, false, true>
      <<<dim3(U2_DIM / 128, B_SAMPLES / 128, K_BLOCKS), 256, 0, stream>>>(
          t1, U1_DIM, Kb2t, U2_DIM, U1_DIM, Bb2, K_BLOCKS, out, U2_DIM, offsets, perm, B_SAMPLES);
}
